// NunchakuLinearAWQ_4947802325523
// MI455X (gfx1250) — matrix operand order repaired and checked by reading, not compiled, not run on hardware
//
#include <hip/hip_runtime.h>
#include <hip/hip_bf16.h>

typedef __attribute__((ext_vector_type(16))) _Float16 v16h;
typedef __attribute__((ext_vector_type(8)))  _Float16 h8;
typedef __attribute__((ext_vector_type(8)))  float    v8f;
typedef __attribute__((ext_vector_type(4)))  int      v4i;

#define IN_F   4096
#define OUT_F  11008
#define GROUPQ 64
#define RANK   32
#define M_TOT  4096      // 2 * 2048 rows of x
#define KT32   32        // k-step per stage
#define NKIT   (IN_F / KT32)
#define LDSB   40        // padded LDS row stride in halves (80B, 16B aligned)

#ifndef __has_builtin
#define __has_builtin(x) 0
#endif

#if __has_builtin(__builtin_amdgcn_global_load_async_to_lds_b128)
#define HAVE_ASYNC 1
#else
#define HAVE_ASYNC 0
#endif

#if __has_builtin(__builtin_amdgcn_s_wait_asynccnt)
#define WAIT_ASYNC(n) __builtin_amdgcn_s_wait_asynccnt(n)
#else
#define WAIT_ASYNC(n) asm volatile("s_wait_asynccnt %0" ::"i"(n) : "memory")
#endif

#define AS1 __attribute__((address_space(1)))
#define AS3 __attribute__((address_space(3)))

__device__ __forceinline__ AS3 v4i* lds_cast4(void* p) {
  return (AS3 v4i*)(unsigned)(unsigned long long)p;
}
__device__ __forceinline__ AS1 v4i* glob_cast4(const void* p) {
  return (AS1 v4i*)(unsigned long long)p;
}

// ---------------------------------------------------------------------------
// Kernel 0: x (f32) -> Xh (f16), one-shot so the GEMM never converts.
// ---------------------------------------------------------------------------
__global__ __launch_bounds__(256) void x_to_f16(const float* __restrict__ x,
                                                _Float16* __restrict__ Xh) {
  const size_t i = ((size_t)blockIdx.x * 256 + threadIdx.x) * 8;
  float4 f0 = *(const float4*)(x + i);
  float4 f1 = *(const float4*)(x + i + 4);
  h8 h = {(_Float16)f0.x, (_Float16)f0.y, (_Float16)f0.z, (_Float16)f0.w,
          (_Float16)f1.x, (_Float16)f1.y, (_Float16)f1.z, (_Float16)f1.w};
  *(h8*)(Xh + i) = h;
}

// ---------------------------------------------------------------------------
// Kernel 1: W_eff[o][i] = q*scale + zero + sum_r lora_up[o][r]*lora_down[i][r]
// (rank-32 LoRA folded into the weights -> single GEMM downstream)
// ---------------------------------------------------------------------------
__global__ __launch_bounds__(256) void awq_dequant_fold(
    const int* __restrict__ qw, const float* __restrict__ scales,
    const float* __restrict__ zeros, const float* __restrict__ ldn,
    const float* __restrict__ lup, _Float16* __restrict__ W) {
  const int tid = threadIdx.x;
  const int o   = blockIdx.x >> 1;
  const int i0  = ((blockIdx.x & 1) << 11) + tid * 8;

  __shared__ float lu[RANK];
  if (tid < RANK) lu[tid] = lup[(size_t)o * RANK + tid];
  __syncthreads();

  const int4* qp = (const int4*)(qw + (size_t)o * IN_F + i0);
  int4 q0 = qp[0];
  int4 q1 = qp[1];
  const int g = i0 >> 6;
  const float s = scales[(size_t)o * (IN_F / GROUPQ) + g];
  const float z = zeros[(size_t)o * (IN_F / GROUPQ) + g];

  float w[8];
  w[0] = (float)q0.x * s + z; w[1] = (float)q0.y * s + z;
  w[2] = (float)q0.z * s + z; w[3] = (float)q0.w * s + z;
  w[4] = (float)q1.x * s + z; w[5] = (float)q1.y * s + z;
  w[6] = (float)q1.z * s + z; w[7] = (float)q1.w * s + z;

#pragma unroll
  for (int j = 0; j < 8; ++j) {
    const float4* ld = (const float4*)(ldn + (size_t)(i0 + j) * RANK);
    float a = 0.f;
#pragma unroll
    for (int r4 = 0; r4 < RANK / 4; ++r4) {
      float4 d = ld[r4];
      const float4 u = *(const float4*)&lu[r4 * 4];
      a += d.x * u.x + d.y * u.y + d.z * u.z + d.w * u.w;
    }
    w[j] += a;
  }

  h8 hv;
#pragma unroll
  for (int j = 0; j < 8; ++j) hv[j] = (_Float16)w[j];
  *(h8*)(W + (size_t)o * IN_F + i0) = hv;
}

// ---------------------------------------------------------------------------
// Kernel 2: out = Xh * W_eff^T + bias. Block tile 128M x 256N, k-step 32,
// ping-pong LDS fed entirely by global_load_async_to_lds_b128.
// 8 waves, each 64x64 -> 16 v_wmma_f32_16x16x32_f16 per k-step.
// ---------------------------------------------------------------------------
__global__ __launch_bounds__(256) void awq_wmma_gemm(
    const _Float16* __restrict__ Xh, const _Float16* __restrict__ W,
    const float* __restrict__ bias, float* __restrict__ out) {
  __shared__ _Float16 At[2][128 * LDSB];  // 20 KB
  __shared__ _Float16 Bt[2][256 * LDSB];  // 40 KB

  const int tid  = threadIdx.x;
  const int lane = tid & 31;
  const int wid  = tid >> 5;
  const int n0   = blockIdx.x * 256;
  const int m0   = blockIdx.y * 128;
  const int mw   = (wid >> 2) * 64;   // 0, 64
  const int nw   = (wid & 3) * 64;    // 0, 64, 128, 192
  const int lrow = lane & 15;
  const int lhi  = lane >> 4;         // 0/1

  // staging assignments
  const int arow = tid >> 1;          // A: 128 rows, 2 threads/row
  const int ah   = tid & 1;           // 32B half of a 64B row

  v8f acc[4][4] = {};

  // issue async loads for tile kt into buffer b
  auto issue_tile = [&](int kt, int b) {
#if HAVE_ASYNC
    {
      const _Float16* ga = Xh + (size_t)(m0 + arow) * IN_F + kt * KT32 + ah * 16;
      _Float16* la = &At[b][arow * LDSB + ah * 16];
      __builtin_amdgcn_global_load_async_to_lds_b128(glob_cast4(ga),
                                                     lds_cast4(la), 0, 0);
      __builtin_amdgcn_global_load_async_to_lds_b128(glob_cast4(ga + 8),
                                                     lds_cast4(la + 8), 0, 0);
    }
    {
      const _Float16* gw = W + (size_t)(n0 + tid) * IN_F + kt * KT32;
      _Float16* lb = &Bt[b][tid * LDSB];
#pragma unroll
      for (int j = 0; j < 4; ++j)
        __builtin_amdgcn_global_load_async_to_lds_b128(
            glob_cast4(gw + j * 8), lds_cast4(lb + j * 8), 0, 0);
    }
#else
    {
      const _Float16* ga = Xh + (size_t)(m0 + arow) * IN_F + kt * KT32 + ah * 16;
      _Float16* la = &At[b][arow * LDSB + ah * 16];
      *(h8*)la = *(const h8*)ga;
      *(h8*)(la + 8) = *(const h8*)(ga + 8);
    }
    {
      const _Float16* gw = W + (size_t)(n0 + tid) * IN_F + kt * KT32;
      _Float16* lb = &Bt[b][tid * LDSB];
#pragma unroll
      for (int j = 0; j < 4; ++j) *(h8*)(lb + j * 8) = *(const h8*)(gw + j * 8);
    }
#endif
  };

  issue_tile(0, 0);

  for (int kt = 0; kt < NKIT; ++kt) {
    __syncthreads();  // all waves done reading buf[(kt+1)&1] (tile kt-1)
    if (kt + 1 < NKIT) {
      issue_tile(kt + 1, (kt + 1) & 1);
      WAIT_ASYNC(6);  // in-order: tile kt's 6 async ops have landed
    } else {
      WAIT_ASYNC(0);
    }
    __syncthreads();  // tile kt visible to all waves

    const int b = kt & 1;
    v16h afr[4], bfr[4];
#pragma unroll
    for (int mf = 0; mf < 4; ++mf) {
      const int ar = mw + mf * 16 + lrow;
      h8 lo = *(const h8*)&At[b][ar * LDSB + lhi * 16];
      h8 hi = *(const h8*)&At[b][ar * LDSB + lhi * 16 + 8];
      afr[mf] = __builtin_shufflevector(lo, hi, 0, 1, 2, 3, 4, 5, 6, 7, 8, 9,
                                        10, 11, 12, 13, 14, 15);
    }
#pragma unroll
    for (int nf = 0; nf < 4; ++nf) {
      const int br = nw + nf * 16 + lrow;
      h8 lo = *(const h8*)&Bt[b][br * LDSB + lhi * 16];
      h8 hi = *(const h8*)&Bt[b][br * LDSB + lhi * 16 + 8];
      bfr[nf] = __builtin_shufflevector(lo, hi, 0, 1, 2, 3, 4, 5, 6, 7, 8, 9,
                                        10, 11, 12, 13, 14, 15);
    }
#pragma unroll
    for (int mf = 0; mf < 4; ++mf)
#pragma unroll
      for (int nf = 0; nf < 4; ++nf)
        acc[mf][nf] = __builtin_amdgcn_wmma_f32_16x16x32_f16(
            false, afr[mf], false, bfr[nf], (short)0, acc[mf][nf], false,
            false);
  }

  // ---- epilogue: + bias, store f32 ----
#pragma unroll
  for (int nf = 0; nf < 4; ++nf) {
    const int col = n0 + nw + nf * 16 + lrow;
    const float bv = bias[col];
#pragma unroll
    for (int mf = 0; mf < 4; ++mf) {
      float* op = out + (size_t)(m0 + mw + mf * 16 + lhi * 8) * OUT_F + col;
      v8f a = acc[mf][nf];
#pragma unroll
      for (int r = 0; r < 8; ++r) op[(size_t)r * OUT_F] = a[r] + bv;
    }
  }
}

// ---------------------------------------------------------------------------
extern "C" void kernel_launch(void* const* d_in, const int* in_sizes, int n_in,
                              void* d_out, int out_size, void* d_ws,
                              size_t ws_size, hipStream_t stream) {
  const float* x      = (const float*)d_in[0];
  const int*   qw     = (const int*)d_in[1];
  const float* scales = (const float*)d_in[2];
  const float* zeros  = (const float*)d_in[3];
  const float* bias   = (const float*)d_in[4];
  const float* ldn    = (const float*)d_in[5];
  const float* lup    = (const float*)d_in[6];
  float*       out    = (float*)d_out;

  _Float16* W  = (_Float16*)d_ws;                       // OUT_F*IN_F f16 (~86 MB)
  _Float16* Xh = W + (size_t)OUT_F * IN_F;              // M_TOT*IN_F f16 (~32 MB)

  x_to_f16<<<(M_TOT * IN_F) / (256 * 8), 256, 0, stream>>>(x, Xh);
  awq_dequant_fold<<<OUT_F * 2, 256, 0, stream>>>(qw, scales, zeros, ldn, lup, W);

  dim3 grid(OUT_F / 256, M_TOT / 128);
  awq_wmma_gemm<<<grid, 256, 0, stream>>>(Xh, W, bias, out);
}
